// Message_Passing4OBJ_33131377721751
// MI455X (gfx1250) — compile-verified, matrix operand order checked by reading, not run on hardware
//
#include <hip/hip_runtime.h>

typedef unsigned short u16;
typedef unsigned int u32;
typedef __attribute__((ext_vector_type(16))) __bf16 v16bf;
typedef __attribute__((ext_vector_type(8)))  float  v8f;

#define NUM_IMG 16
#define N_PER   128
#define R_PER   1024
#define D_DIM   512
#define N_REL   (NUM_IMG * R_PER)   // 16384
#define N_OBJ   (NUM_IMG * N_PER)   // 2048

union Frag { v16bf v; uint4 q[2]; };

__device__ __forceinline__ u16 f2bf(float f) {
    u32 u = __float_as_uint(f);
    u32 r = (u + 0x7FFFu + ((u >> 16) & 1u)) >> 16;
    return (u16)r;
}
__device__ __forceinline__ float bf2f(u16 h) {
    return __uint_as_float(((u32)h) << 16);
}
__device__ __forceinline__ v8f vzero() {
    v8f z;
#pragma unroll
    for (int i = 0; i < 8; i++) z[i] = 0.0f;
    return z;
}

// CDNA5 async global->LDS copy (16B), ASYNCcnt-tracked.
// INST_OFFSET applies to BOTH the LDS and global addresses, so consecutive
// 16B chunks reuse one base VGPR (no address VALU).
__device__ __forceinline__ u32 lds_off(const void* p) { return (u32)(size_t)p; }

#define ASYNC_G2L(loff, gptr, OFF)                                            \
    asm volatile("global_load_async_to_lds_b128 %0, %1, off offset:" #OFF     \
                 :: "v"(loff), "v"(gptr) : "memory")

__device__ __forceinline__ void wait_asynccnt0() {
    asm volatile("s_wait_asynccnt 0x0" ::: "memory");
}

// ---------------------------------------------------------------- cvt f32->bf16
__global__ void cvt_kernel(const float* __restrict__ src, u16* __restrict__ dst, int n) {
    int i = blockIdx.x * blockDim.x + threadIdx.x;
    if (i < n) dst[i] = f2bf(src[i]);
}

// cvt + transpose: src[K,N] f32 -> dst[N,K] bf16
__global__ void cvt_t_kernel(const float* __restrict__ src, u16* __restrict__ dst,
                             int K, int N) {
    int i = blockIdx.x * blockDim.x + threadIdx.x;
    if (i >= K * N) return;
    int k = i / N, n = i - k * N;
    dst[(size_t)n * K + k] = f2bf(src[i]);
}

__global__ void zero_kernel(float* __restrict__ p, int n) {
    int i = blockIdx.x * blockDim.x + threadIdx.x;
    if (i < n) p[i] = 0.0f;
}

// ---------------------------------------------------------------- generic bf16 WMMA GEMM
// C[M,N] = gather(A)[M,K] @ Bt[N,K]^T + bias ; optional relu; f32 and/or bf16 out.
// Block tile 128x128, 8 waves (4x2), each wave 32x64 (2x4 wmma), K-step 32.
// Double-buffered LDS, async global->LDS staging with immediate offsets.
__global__ __launch_bounds__(256) void gemm_bf16_kernel(
    const u16* __restrict__ A, int lda,
    const int* __restrict__ gidx, int gstride,
    const u16* __restrict__ Bt, int N, int K,
    const float* __restrict__ bias,
    float* __restrict__ outF, u16* __restrict__ outH, int relu)
{
    __shared__ __align__(16) u16 As[2][128 * 40];
    __shared__ __align__(16) u16 Bs[2][128 * 40];

    const int tid  = threadIdx.x;
    const int lane = tid & 31, wave = tid >> 5;
    const int wm = wave >> 1, wn = wave & 1;
    const int lm = lane & 15, koff = (lane >> 4) << 4;
    const int m0 = blockIdx.y * 128, n0 = blockIdx.x * 128;

    // staging coords: 2 threads per row, 16 halves (32B) each
    const int r    = tid >> 1;
    const int part = (tid & 1) << 4;
    int arow = m0 + r;
    if (gidx) arow = gidx[(size_t)arow * gstride];
    const u16* acur = A  + (size_t)arow * lda + part;
    const u16* bcur = Bt + (size_t)(n0 + r) * K + part;
    const u32 asOff0 = lds_off(&As[0][r * 40 + part]);
    const u32 bsOff0 = lds_off(&Bs[0][r * 40 + part]);
    const u32 asOff1 = lds_off(&As[1][r * 40 + part]);
    const u32 bsOff1 = lds_off(&Bs[1][r * 40 + part]);

    v8f acc[2][4];
#pragma unroll
    for (int i = 0; i < 2; i++)
#pragma unroll
        for (int j = 0; j < 4; j++) acc[i][j] = vzero();

    const int nk = K >> 5;

    auto stage = [&](int buf) {
        const u32 as = buf ? asOff1 : asOff0;
        const u32 bs = buf ? bsOff1 : bsOff0;
        ASYNC_G2L(as, acur, 0);
        ASYNC_G2L(as, acur, 16);
        ASYNC_G2L(bs, bcur, 0);
        ASYNC_G2L(bs, bcur, 16);
        acur += 32; bcur += 32;
        __builtin_prefetch(acur, 0, 0);
        __builtin_prefetch(bcur, 0, 0);
    };

    stage(0);
    wait_asynccnt0();
    __syncthreads();

    for (int kt = 0; kt < nk; kt++) {
        const int buf = kt & 1;
        if (kt + 1 < nk) stage(buf ^ 1);

        Frag a[2], b[4];
#pragma unroll
        for (int mi = 0; mi < 2; mi++) {
            const uint4* s = (const uint4*)&As[buf][(wm * 32 + mi * 16 + lm) * 40 + koff];
            a[mi].q[0] = s[0]; a[mi].q[1] = s[1];
        }
#pragma unroll
        for (int ni = 0; ni < 4; ni++) {
            const uint4* s = (const uint4*)&Bs[buf][(wn * 64 + ni * 16 + lm) * 40 + koff];
            b[ni].q[0] = s[0]; b[ni].q[1] = s[1];
        }
#pragma unroll
        for (int mi = 0; mi < 2; mi++)
#pragma unroll
            for (int ni = 0; ni < 4; ni++)
                acc[mi][ni] = __builtin_amdgcn_wmma_f32_16x16x32_bf16(
                    false, a[mi].v, false, b[ni].v, (short)0, acc[mi][ni], false, false);

        wait_asynccnt0();
        __syncthreads();
    }

    const int em = (lane >> 4) << 3, en = lane & 15;
#pragma unroll
    for (int mi = 0; mi < 2; mi++)
#pragma unroll
        for (int ni = 0; ni < 4; ni++) {
            int gm0 = m0 + wm * 32 + mi * 16 + em;
            int gn  = n0 + wn * 64 + ni * 16 + en;
            float bv = bias ? bias[gn] : 0.0f;
#pragma unroll
            for (int rr = 0; rr < 8; rr++) {
                float v = acc[mi][ni][rr] + bv;
                if (relu) v = fmaxf(v, 0.0f);
                size_t idx = (size_t)(gm0 + rr) * N + gn;
                if (outF) outF[idx] = v;
                if (outH) outH[idx] = f2bf(v);
            }
        }
}

// ---------------------------------------------------------------- fused fusion-mid
// O[M,256] = sum_{ch=0..14} (U@W0[:,ch] + b0[ch]) * (V@W1[:,ch] + b1[ch])
// W0t/W1t pre-transposed: [3840, 256].
// Block: 32 rows, all 256 cols; 8 waves each own a 32-col strip (2x2 wmma tiles).
__global__ __launch_bounds__(256) void fusion_mid_kernel(
    const u16* __restrict__ U, const u16* __restrict__ V,
    const u16* __restrict__ W0t, const u16* __restrict__ W1t,
    const float* __restrict__ bias0, const float* __restrict__ bias1,
    u16* __restrict__ O)
{
    __shared__ __align__(16) u16 Us[32 * 264];
    __shared__ __align__(16) u16 Vs[32 * 264];
    __shared__ __align__(16) u16 Ws[256 * 40];

    const int tid  = threadIdx.x;
    const int lane = tid & 31, wave = tid >> 5;
    const int lm = lane & 15, koff = (lane >> 4) << 4;
    const int m0 = blockIdx.x * 32;

    {   // stage U,V tiles (32 x 256) once via async copies; reused across all 15 chunks
        const int r  = tid >> 3;
        const int c0 = (tid & 7) << 5;
        const u16* gu = U + (size_t)(m0 + r) * 256 + c0;
        const u16* gv = V + (size_t)(m0 + r) * 256 + c0;
        const u32 su = lds_off(&Us[r * 264 + c0]);
        const u32 sv = lds_off(&Vs[r * 264 + c0]);
        ASYNC_G2L(su, gu, 0);
        ASYNC_G2L(su, gu, 16);
        ASYNC_G2L(su, gu, 32);
        ASYNC_G2L(su, gu, 48);
        ASYNC_G2L(sv, gv, 0);
        ASYNC_G2L(sv, gv, 16);
        ASYNC_G2L(sv, gv, 32);
        ASYNC_G2L(sv, gv, 48);
        wait_asynccnt0();
    }

    v8f oacc[2][2];
#pragma unroll
    for (int i = 0; i < 2; i++)
#pragma unroll
        for (int j = 0; j < 2; j++) oacc[i][j] = vzero();

    // W staging: thread t owns transposed row (n-local == t), 32 halves per k-step
    const u16* w0p = W0t + (size_t)tid * 256;
    const u16* w1p = W1t + (size_t)tid * 256;
    const u32 wsOff = lds_off(&Ws[tid * 40]);
    const int em = (lane >> 4) << 3, en = lane & 15;

    for (int ch = 0; ch < 15; ch++) {
        v8f aacc[2][2], bacc[2][2];
#pragma unroll
        for (int i = 0; i < 2; i++)
#pragma unroll
            for (int j = 0; j < 2; j++) { aacc[i][j] = vzero(); bacc[i][j] = vzero(); }

        const size_t chOff = (size_t)ch * 256 * 256;

        // ---- pass 0: U @ W0 chunk ----
        for (int ks = 0; ks < 8; ks++) {
            __syncthreads();
            {
                const u16* g = w0p + chOff + (ks << 5);
                ASYNC_G2L(wsOff, g, 0);
                ASYNC_G2L(wsOff, g, 16);
                ASYNC_G2L(wsOff, g, 32);
                ASYNC_G2L(wsOff, g, 48);
                wait_asynccnt0();
            }
            __syncthreads();
            Frag a[2], b[2];
#pragma unroll
            for (int mi = 0; mi < 2; mi++) {
                const uint4* s = (const uint4*)&Us[(mi * 16 + lm) * 264 + (ks << 5) + koff];
                a[mi].q[0] = s[0]; a[mi].q[1] = s[1];
            }
#pragma unroll
            for (int ni = 0; ni < 2; ni++) {
                const uint4* s = (const uint4*)&Ws[(wave * 32 + ni * 16 + lm) * 40 + koff];
                b[ni].q[0] = s[0]; b[ni].q[1] = s[1];
            }
#pragma unroll
            for (int mi = 0; mi < 2; mi++)
#pragma unroll
                for (int ni = 0; ni < 2; ni++)
                    aacc[mi][ni] = __builtin_amdgcn_wmma_f32_16x16x32_bf16(
                        false, a[mi].v, false, b[ni].v, (short)0, aacc[mi][ni], false, false);
        }

        // ---- pass 1: V @ W1 chunk ----
        for (int ks = 0; ks < 8; ks++) {
            __syncthreads();
            {
                const u16* g = w1p + chOff + (ks << 5);
                ASYNC_G2L(wsOff, g, 0);
                ASYNC_G2L(wsOff, g, 16);
                ASYNC_G2L(wsOff, g, 32);
                ASYNC_G2L(wsOff, g, 48);
                wait_asynccnt0();
            }
            __syncthreads();
            Frag a[2], b[2];
#pragma unroll
            for (int mi = 0; mi < 2; mi++) {
                const uint4* s = (const uint4*)&Vs[(mi * 16 + lm) * 264 + (ks << 5) + koff];
                a[mi].q[0] = s[0]; a[mi].q[1] = s[1];
            }
#pragma unroll
            for (int ni = 0; ni < 2; ni++) {
                const uint4* s = (const uint4*)&Ws[(wave * 32 + ni * 16 + lm) * 40 + koff];
                b[ni].q[0] = s[0]; b[ni].q[1] = s[1];
            }
#pragma unroll
            for (int mi = 0; mi < 2; mi++)
#pragma unroll
                for (int ni = 0; ni < 2; ni++)
                    bacc[mi][ni] = __builtin_amdgcn_wmma_f32_16x16x32_bf16(
                        false, a[mi].v, false, b[ni].v, (short)0, bacc[mi][ni], false, false);
        }

        // ---- combine: o += (a + b0) * (b + b1) ----
#pragma unroll
        for (int mi = 0; mi < 2; mi++)
#pragma unroll
            for (int ni = 0; ni < 2; ni++) {
                int gn = ch * 256 + wave * 32 + ni * 16 + en;
                float c0b = bias0[gn], c1b = bias1[gn];
#pragma unroll
                for (int rr = 0; rr < 8; rr++)
                    oacc[mi][ni][rr] += (aacc[mi][ni][rr] + c0b) * (bacc[mi][ni][rr] + c1b);
            }
    }

#pragma unroll
    for (int mi = 0; mi < 2; mi++)
#pragma unroll
        for (int ni = 0; ni < 2; ni++)
#pragma unroll
            for (int rr = 0; rr < 8; rr++)
                O[(size_t)(m0 + mi * 16 + em + rr) * 256 + wave * 32 + ni * 16 + en] =
                    f2bf(oacc[mi][ni][rr]);
}

// ---------------------------------------------------------------- fold wo@w_W
__global__ __launch_bounds__(256) void fusew_kernel(
    const float* __restrict__ wo, const float* __restrict__ bo,
    const float* __restrict__ wW, const float* __restrict__ wb,
    float* __restrict__ wv, float* __restrict__ cc)
{
    int j = threadIdx.x;
    float s = 0.0f;
    for (int d = 0; d < 512; d++) s += wo[j * 512 + d] * wW[d];
    wv[j] = s;
    if (j == 0) {
        float c = wb[0];
        for (int d = 0; d < 512; d++) c += bo[d] * wW[d];
        cc[0] = c;
    }
}

// ---------------------------------------------------------------- att = o2.wv + c
__global__ void att_kernel(const u16* __restrict__ o2, const float* __restrict__ wv,
                           const float* __restrict__ cc, float* __restrict__ att)
{
    int r = blockIdx.x * blockDim.x + threadIdx.x;
    if (r >= N_REL) return;
    float s = cc[0];
    const u16* row = o2 + (size_t)r * 256;
    for (int j = 0; j < 256; j++) s += bf2f(row[j]) * wv[j];
    att[r] = s;
}

// ---------------------------------------------------------------- scatter att into A
__global__ void buildA_kernel(const int* __restrict__ rel, const float* __restrict__ att,
                              float* __restrict__ A)
{
    int r = blockIdx.x * blockDim.x + threadIdx.x;
    if (r >= N_REL) return;
    int i = rel[r * 3 + 0];
    int s = rel[r * 3 + 1] - i * N_PER;
    int o = rel[r * 3 + 2] - i * N_PER;
    atomicAdd(&A[(size_t)i * N_PER * N_PER + s * N_PER + o], att[r]);
}

// ---------------------------------------------------------------- sigmoid, mask, normalize
__global__ __launch_bounds__(128) void normA_kernel(float* __restrict__ A)
{
    __shared__ float rs[N_PER];
    int img = blockIdx.x, t = threadIdx.x;
    float* Ai = A + (size_t)img * N_PER * N_PER;
    float sum = 0.0f;
    for (int j = 0; j < N_PER; j++) {
        float v = Ai[t * N_PER + j];
        v = 1.0f / (1.0f + __expf(-v));
        if (j == t) v = 0.0f;
        Ai[t * N_PER + j] = v;
        sum += v;
    }
    rs[t] = sum;
    __syncthreads();
    for (int j = 0; j < N_PER; j++) Ai[t * N_PER + j] /= rs[j];
}

// ---------------------------------------------------------------- ctx = [A@C ; A^T@C]
__global__ void ctx_kernel(const float* __restrict__ A, const float* __restrict__ C,
                           float* __restrict__ ctx)
{
    int t = blockIdx.x * blockDim.x + threadIdx.x;   // 0 .. 2048*512-1
    int m = t >> 9, n = t & 511;
    int img = m >> 7, ml = m & 127;
    const float* Ai = A + (size_t)img * N_PER * N_PER;
    const float* Ci = C + (size_t)img * N_PER * 256;
    float s = 0.0f;
    if (n < 256) {
        for (int k = 0; k < N_PER; k++) s += Ai[ml * N_PER + k] * Ci[k * 256 + n];
    } else {
        int nn = n - 256;
        for (int k = 0; k < N_PER; k++) s += Ai[k * N_PER + ml] * Ci[k * 256 + nn];
    }
    ctx[(size_t)m * 512 + n] = s;
}

// ---------------------------------------------------------------- fused LN head
__global__ __launch_bounds__(128) void head_kernel(
    const float* __restrict__ ctx,
    const float* __restrict__ tW1, const float* __restrict__ tb1,
    const float* __restrict__ tg,  const float* __restrict__ tbe,
    const float* __restrict__ tW2, const float* __restrict__ tb2,
    const float* __restrict__ obj, float* __restrict__ out)
{
    __shared__ float cs[512];
    __shared__ float hs[128];
    __shared__ float red[128];
    int m = blockIdx.x, t = threadIdx.x;
    for (int j = t; j < 512; j += 128) cs[j] = ctx[(size_t)m * 512 + j];
    __syncthreads();
    float h = tb1[t];
    for (int d = 0; d < 512; d++) h += cs[d] * tW1[d * 128 + t];
    red[t] = h;
    __syncthreads();
    for (int s = 64; s > 0; s >>= 1) { if (t < s) red[t] += red[t + s]; __syncthreads(); }
    float mu = red[0] / 128.0f;
    __syncthreads();
    float dv = h - mu;
    red[t] = dv * dv;
    __syncthreads();
    for (int s = 64; s > 0; s >>= 1) { if (t < s) red[t] += red[t + s]; __syncthreads(); }
    float var = red[0] / 128.0f;
    float hn = dv * rsqrtf(var + 1e-5f) * tg[t] + tbe[t];
    hs[t] = fmaxf(hn, 0.0f);
    __syncthreads();
    for (int q = 0; q < 4; q++) {
        int n = t + q * 128;
        float v = tb2[n];
        for (int k = 0; k < 128; k++) v += hs[k] * tW2[k * 512 + n];
        v += obj[(size_t)m * 512 + n];
        out[(size_t)m * 512 + n] = fmaxf(v, 0.0f);
    }
}

// ================================================================ launch
extern "C" void kernel_launch(void* const* d_in, const int* in_sizes, int n_in,
                              void* d_out, int out_size, void* d_ws, size_t ws_size,
                              hipStream_t stream) {
    (void)in_sizes; (void)n_in; (void)out_size; (void)ws_size;
    const float* obj    = (const float*)d_in[0];
    const float* phr    = (const float*)d_in[1];
    const int*   rel    = (const int*)d_in[3];
    const float* f1_w0a = (const float*)d_in[4];  const float* f1_b0a = (const float*)d_in[5];
    const float* f1_w0b = (const float*)d_in[6];  const float* f1_b0b = (const float*)d_in[7];
    const float* f1_w1a = (const float*)d_in[8];  const float* f1_b1a = (const float*)d_in[9];
    const float* f1_w1b = (const float*)d_in[10]; const float* f1_b1b = (const float*)d_in[11];
    const float* f1_wo  = (const float*)d_in[12]; const float* f1_bo  = (const float*)d_in[13];
    const float* f2_w0a = (const float*)d_in[14]; const float* f2_b0a = (const float*)d_in[15];
    const float* f2_w0b = (const float*)d_in[16]; const float* f2_b0b = (const float*)d_in[17];
    const float* f2_w1a = (const float*)d_in[18]; const float* f2_b1a = (const float*)d_in[19];
    const float* f2_w1b = (const float*)d_in[20]; const float* f2_b1b = (const float*)d_in[21];
    const float* f2_wo  = (const float*)d_in[22]; const float* f2_bo  = (const float*)d_in[23];
    const float* w_W    = (const float*)d_in[24]; const float* w_b    = (const float*)d_in[25];
    const float* conv_W = (const float*)d_in[26]; const float* conv_b = (const float*)d_in[27];
    const float* t_W1   = (const float*)d_in[28]; const float* t_b1   = (const float*)d_in[29];
    const float* t_g    = (const float*)d_in[30]; const float* t_be   = (const float*)d_in[31];
    const float* t_W2   = (const float*)d_in[32]; const float* t_b2   = (const float*)d_in[33];
    float* out = (float*)d_out;

    char* ws = (char*)d_ws;
    size_t off = 0;
    auto alloc = [&](size_t b) -> char* {
        char* p = ws + off;
        off = (off + b + 255) & ~(size_t)255;
        return p;
    };

    u16* obj_h    = (u16*)alloc((size_t)N_OBJ * D_DIM * 2);
    u16* phr_h    = (u16*)alloc((size_t)N_REL * D_DIM * 2);
    u16* f1w0a_t  = (u16*)alloc(512 * 256 * 2);
    u16* f1w1a_t  = (u16*)alloc(512 * 256 * 2);
    u16* f1w0b_t  = (u16*)alloc((size_t)256 * 3840 * 2);
    u16* f1w1b_t  = (u16*)alloc((size_t)256 * 3840 * 2);
    u16* f1wo_t   = (u16*)alloc(256 * 512 * 2);
    u16* f2w0a_t  = (u16*)alloc(512 * 256 * 2);
    u16* f2w1a_t  = (u16*)alloc(512 * 256 * 2);
    u16* f2w0b_t  = (u16*)alloc((size_t)256 * 3840 * 2);
    u16* f2w1b_t  = (u16*)alloc((size_t)256 * 3840 * 2);
    u16* convW_t  = (u16*)alloc(512 * 256 * 2);
    u16* u1       = (u16*)alloc((size_t)N_REL * 256 * 2);
    u16* v1       = (u16*)alloc((size_t)N_REL * 256 * 2);
    u16* o1       = (u16*)alloc((size_t)N_REL * 256 * 2);
    u16* prod_h   = (u16*)alloc((size_t)N_REL * 512 * 2);
    u16* u2       = (u16*)alloc((size_t)N_REL * 256 * 2);
    u16* v2       = (u16*)alloc((size_t)N_REL * 256 * 2);
    u16* o2       = (u16*)alloc((size_t)N_REL * 256 * 2);
    float* wv     = (float*)alloc(256 * 4);
    float* cc     = (float*)alloc(256);
    float* att    = (float*)alloc((size_t)N_REL * 4);
    float* Amat   = (float*)alloc((size_t)NUM_IMG * N_PER * N_PER * 4);
    float* Cmat   = (float*)alloc((size_t)N_OBJ * 256 * 4);
    float* ctx    = (float*)alloc((size_t)N_OBJ * 512 * 4);

    auto cvt = [&](const float* s, u16* d, int n) {
        cvt_kernel<<<(n + 255) / 256, 256, 0, stream>>>(s, d, n);
    };
    auto cvtT = [&](const float* s, u16* d, int K, int N) {
        cvt_t_kernel<<<(K * N + 255) / 256, 256, 0, stream>>>(s, d, K, N);
    };
    cvt(obj, obj_h, N_OBJ * D_DIM);
    cvt(phr, phr_h, N_REL * D_DIM);
    cvtT(f1_w0a, f1w0a_t, 512, 256);
    cvtT(f1_w1a, f1w1a_t, 512, 256);
    cvtT(f1_w0b, f1w0b_t, 256, 3840);
    cvtT(f1_w1b, f1w1b_t, 256, 3840);
    cvtT(f1_wo,  f1wo_t,  256, 512);
    cvtT(f2_w0a, f2w0a_t, 512, 256);
    cvtT(f2_w1a, f2w1a_t, 512, 256);
    cvtT(f2_w0b, f2w0b_t, 256, 3840);
    cvtT(f2_w1b, f2w1b_t, 256, 3840);
    cvtT(conv_W, convW_t, 512, 256);

    // ---- fusion 1: prod = fusion(obj[s], obj[o]) ----
    gemm_bf16_kernel<<<dim3(256 / 128, N_REL / 128), 256, 0, stream>>>(
        obj_h, 512, rel + 1, 3, f1w0a_t, 256, 512, f1_b0a, nullptr, u1, 0);
    gemm_bf16_kernel<<<dim3(256 / 128, N_REL / 128), 256, 0, stream>>>(
        obj_h, 512, rel + 2, 3, f1w1a_t, 256, 512, f1_b1a, nullptr, v1, 0);
    fusion_mid_kernel<<<N_REL / 32, 256, 0, stream>>>(
        u1, v1, f1w0b_t, f1w1b_t, f1_b0b, f1_b1b, o1);
    gemm_bf16_kernel<<<dim3(512 / 128, N_REL / 128), 256, 0, stream>>>(
        o1, 256, nullptr, 0, f1wo_t, 512, 256, f1_bo, nullptr, prod_h, 0);

    // ---- fusion 2: o2 (pair pre-projection), att folded through wo/w_W ----
    gemm_bf16_kernel<<<dim3(256 / 128, N_REL / 128), 256, 0, stream>>>(
        prod_h, 512, nullptr, 0, f2w0a_t, 256, 512, f2_b0a, nullptr, u2, 0);
    gemm_bf16_kernel<<<dim3(256 / 128, N_REL / 128), 256, 0, stream>>>(
        phr_h, 512, nullptr, 0, f2w1a_t, 256, 512, f2_b1a, nullptr, v2, 0);
    fusion_mid_kernel<<<N_REL / 32, 256, 0, stream>>>(
        u2, v2, f2w0b_t, f2w1b_t, f2_b0b, f2_b1b, o2);

    fusew_kernel<<<1, 256, 0, stream>>>(f2_wo, f2_bo, w_W, w_b, wv, cc);
    att_kernel<<<N_REL / 256, 256, 0, stream>>>(o2, wv, cc, att);

    // ---- attention matrix per image ----
    zero_kernel<<<(NUM_IMG * N_PER * N_PER + 255) / 256, 256, 0, stream>>>(
        Amat, NUM_IMG * N_PER * N_PER);
    buildA_kernel<<<N_REL / 256, 256, 0, stream>>>(rel, att, Amat);
    normA_kernel<<<NUM_IMG, 128, 0, stream>>>(Amat);

    // ---- C = relu(obj @ conv_W + conv_b) ----
    gemm_bf16_kernel<<<dim3(256 / 128, N_OBJ / 128), 256, 0, stream>>>(
        obj_h, 512, nullptr, 0, convW_t, 256, 512, conv_b, Cmat, nullptr, 1);

    // ---- ctx = [A@C ; A^T@C], then fused LN head ----
    ctx_kernel<<<(N_OBJ * 512) / 256, 256, 0, stream>>>(Amat, Cmat, ctx);
    head_kernel<<<N_OBJ, 128, 0, stream>>>(
        ctx, t_W1, t_b1, t_g, t_be, t_W2, t_b2, obj, out);
}
